// GAUBlock_17901423689730
// MI455X (gfx1250) — matrix operand order repaired and checked by reading, not compiled, not run on hardware
//
#include <hip/hip_runtime.h>

typedef _Float16 v8h  __attribute__((ext_vector_type(8)));
typedef _Float16 v16h __attribute__((ext_vector_type(16)));
typedef float    v8f  __attribute__((ext_vector_type(8)));

#define DIM    256
#define LSEQ   4096
#define NB     4
#define BLROWS (NB * LSEQ)   // 16384
#define QKD    128
#define HIDN   256

__device__ __forceinline__ v8f wmma32(v16h a, v16h b, v8f c) {
  // D = A(16x32 f16) x B(32x16 f16) + C(16x16 f32)
  return __builtin_amdgcn_wmma_f32_16x16x32_f16(false, a, false, b, (short)0, c,
                                                false, false);
}
__device__ __forceinline__ v16h pack16(v8h lo, v8h hi) {
  v16h r;
#pragma unroll
  for (int i = 0; i < 8; ++i) { r[i] = lo[i]; r[i + 8] = hi[i]; }
  return r;
}
__device__ __forceinline__ float silu_f(float x) { return x / (1.0f + __expf(-x)); }

// ---------------------------------------------------------------------------
// Kernel 1: emb[b,d] = t[b,:] . W_embed[d,:] + b_embed[d]        (tiny GEMV)
// ---------------------------------------------------------------------------
__global__ __launch_bounds__(256) void emb_kernel(const float* __restrict__ t,
                                                  const float* __restrict__ We,
                                                  const float* __restrict__ be,
                                                  float* __restrict__ emb) {
  const int b = blockIdx.x, d = threadIdx.x;
  const float* tr = t + (size_t)b * 3 * DIM;
  const float* wr = We + (size_t)d * 3 * DIM;
  float s = be[d];
  for (int k = 0; k < 3 * DIM; ++k) s += tr[k] * wr[k];
  emb[b * DIM + d] = s;
}

// ---------------------------------------------------------------------------
// Kernel 2: normed[b*L+l, d] = LN_d(x[b,d,l] + emb[b,d]) * g[d] + b[d]  (f16)
// 32 l-rows per block, LDS transpose for coalesced global reads along L.
// ---------------------------------------------------------------------------
__global__ __launch_bounds__(256) void ln_kernel(const float* __restrict__ x,
                                                 const float* __restrict__ emb,
                                                 const float* __restrict__ g,
                                                 const float* __restrict__ bb,
                                                 _Float16* __restrict__ normed) {
  __shared__ float tr[32 * 261];
  const int tid = threadIdx.x;
  const int b = blockIdx.y;
  const int l0 = blockIdx.x * 32;
  {
    const int ll = tid & 31, doff = tid >> 5;
#pragma unroll 8
    for (int it = 0; it < 32; ++it) {
      const int d = it * 8 + doff;
      tr[ll * 261 + d] = x[((size_t)b * DIM + d) * LSEQ + l0 + ll] + emb[b * DIM + d];
    }
  }
  __syncthreads();
  const int row = tid >> 3, part = tid & 7;
  float vals[32];
  float s = 0.f, s2 = 0.f;
#pragma unroll
  for (int j = 0; j < 32; ++j) {
    const float v = tr[row * 261 + part + 8 * j];
    vals[j] = v; s += v; s2 += v * v;
  }
#pragma unroll
  for (int off = 1; off < 8; off <<= 1) {
    s  += __shfl_xor(s,  off, 32);
    s2 += __shfl_xor(s2, off, 32);
  }
  const float mu   = s * (1.0f / 256.0f);
  const float var  = s2 * (1.0f / 256.0f) - mu * mu;
  const float rstd = rsqrtf(var + 1e-5f);
  _Float16* out = normed + ((size_t)(b * LSEQ + l0 + row)) * DIM;
#pragma unroll
  for (int j = 0; j < 32; ++j) {
    const int d = part + 8 * j;
    out[d] = (_Float16)((vals[j] - mu) * rstd * g[d] + bb[d]);
  }
}

// ---------------------------------------------------------------------------
// WMMA GEMM: C(M=16384, N) = A(M,256) f16 @ W(N,256)^T f32 + bias, + epilogue
// Block tile 128x64, 8 waves. A-fragments loaded straight from global (layout
// matches row-major; each element read exactly once per block). W tile staged
// once in LDS (f32->f16), reused by all 8 waves. Inner k-loop is software-
// pipelined with double-buffered fragment registers so LDS loads of step k+1
// overlap the 4 wmma of step k.
// MODE 0: SiLU -> vT/uT (transposed stores)     (N=512)
// MODE 1: SiLU, gamma/beta, RoPE -> q,k         (N=128)
// MODE 2: +bias -> outbl f16                    (N=256)
// MODE 3: +bias; fused residual/skip transpose  (N=512)
// ---------------------------------------------------------------------------
template <int MODE>
__global__ __launch_bounds__(256) void gemm256(
    const _Float16* __restrict__ A, const float* __restrict__ W,
    const float* __restrict__ bias, _Float16* __restrict__ oA,
    _Float16* __restrict__ oB, const float* __restrict__ e0,
    const float* __restrict__ e1, float* __restrict__ f0,
    float* __restrict__ f1) {
  __shared__ _Float16 sW[64 * 264];

  const int tid = threadIdx.x;
  const int m0 = blockIdx.x * 128;
  const int n0 = blockIdx.y * 64;
  const int w = tid >> 5, lane = tid & 31, h = lane >> 4, ln = lane & 15;

  // stage W tile (64 x 256, f32 -> f16), single barrier
#pragma unroll
  for (int it = 0; it < 16; ++it) {
    const int idx = it * 1024 + tid * 4;
    const int row = idx >> 8, col = idx & 255;
    const float4 wv = *(const float4*)(W + (size_t)(n0 + row) * 256 + col);
    _Float16* dst = sW + row * 264 + col;
    dst[0] = (_Float16)wv.x; dst[1] = (_Float16)wv.y;
    dst[2] = (_Float16)wv.z; dst[3] = (_Float16)wv.w;
  }
  __syncthreads();

  const v8f vzero = {0.f, 0.f, 0.f, 0.f, 0.f, 0.f, 0.f, 0.f};
  v8f acc[4];
#pragma unroll
  for (int i = 0; i < 4; ++i) acc[i] = vzero;

  const _Float16* arow = A + (size_t)(m0 + w * 16 + ln) * 256;
  const _Float16* brow = sW + ln * 264 + h * 8;

  v16h afb[2];
  v16h bfb[2][4];
  afb[0] = pack16(*(const v8h*)(arow + h * 8), *(const v8h*)(arow + 16 + h * 8));
#pragma unroll
  for (int nt = 0; nt < 4; ++nt) {
    const _Float16* bp = brow + nt * 16 * 264;
    bfb[0][nt] = pack16(*(const v8h*)bp, *(const v8h*)(bp + 16));
  }
#pragma unroll
  for (int k0 = 0; k0 < 8; ++k0) {
    const int cur = k0 & 1, nxt = cur ^ 1;
    if (k0 < 7) {  // prefetch next step's fragments into alternate buffer
      const _Float16* ap = arow + (k0 + 1) * 32;
      afb[nxt] = pack16(*(const v8h*)(ap + h * 8), *(const v8h*)(ap + 16 + h * 8));
#pragma unroll
      for (int nt = 0; nt < 4; ++nt) {
        const _Float16* bp = brow + nt * 16 * 264 + (k0 + 1) * 32;
        bfb[nxt][nt] = pack16(*(const v8h*)bp, *(const v8h*)(bp + 16));
      }
    }
#pragma unroll
    for (int nt = 0; nt < 4; ++nt)
      acc[nt] = wmma32(afb[cur], bfb[cur][nt], acc[nt]);
  }

  const int rowb = m0 + w * 16 + 8 * h;  // C rows rowb..rowb+7 (M = r + 8h)

  if (MODE == 0) {  // hv: SiLU, split v/u, transposed f16 stores (16B each)
#pragma unroll
    for (int nt = 0; nt < 4; ++nt) {
      const int c = n0 + nt * 16 + ln;
      const float bi = bias[c];
      v8h hv;
#pragma unroll
      for (int r = 0; r < 8; ++r) hv[r] = (_Float16)silu_f(acc[nt][r] + bi);
      if (c < HIDN)
        *(v8h*)(oA + (size_t)c * BLROWS + rowb) = hv;
      else
        *(v8h*)(oB + (size_t)(c - HIDN) * BLROWS + rowb) = hv;
    }
  } else if (MODE == 1) {  // Z: SiLU, gamma/beta, RoPE -> q, k
    const float LG = 9.210340372f / 128.0f;  // ln(10000)/d
#pragma unroll
    for (int nt = 0; nt < 4; ++nt) {
      const int c = n0 + nt * 16 + ln;  // 0..127
      const float bi = bias[c];
      const float g0 = e0[c], g1 = e0[QKD + c];
      const float bt0 = e1[c], bt1 = e1[QKD + c];
      const float fr = __expf(-(float)(c & ~1) * LG);
      const bool even = (c & 1) == 0;
#pragma unroll
      for (int r = 0; r < 8; ++r) {
        const int row = rowb + r;
        const float z = silu_f(acc[nt][r] + bi);
        const float zq = z * g0 + bt0;
        const float zk = z * g1 + bt1;
        const float zqp = __shfl_xor(zq, 1, 32);  // paired cols = adjacent lanes
        const float zkp = __shfl_xor(zk, 1, 32);
        const float ang = (float)(row & (LSEQ - 1)) * fr;
        float sn, cs;
        __sincosf(ang, &sn, &cs);
        const float qv = even ? (zq * cs - zqp * sn) : (zqp * sn + zq * cs);
        const float kv = even ? (zk * cs - zkp * sn) : (zkp * sn + zk * cs);
        oA[(size_t)row * QKD + c] = (_Float16)qv;
        oB[(size_t)row * QKD + c] = (_Float16)kv;
      }
    }
  } else if (MODE == 2) {  // out GEMM -> row-major f16
#pragma unroll
    for (int nt = 0; nt < 4; ++nt) {
      const int c = n0 + nt * 16 + ln;
      const float bi = bias[c];
#pragma unroll
      for (int r = 0; r < 8; ++r)
        oA[(size_t)(rowb + r) * HIDN + c] = (_Float16)(acc[nt][r] + bi);
    }
  } else {  // proj: fused transpose + residual/skip (float4 stores along L)
#pragma unroll
    for (int nt = 0; nt < 4; ++nt) {
      const int c = n0 + nt * 16 + ln;  // 0..511
      const float bi = bias[c];
      const int b = rowb >> 12, l = rowb & (LSEQ - 1);
      if (c < DIM) {
        const float* xp = e0 + ((size_t)b * DIM + c) * LSEQ + l;
        const float ev = e1[b * DIM + c];
        float* op = f0 + ((size_t)b * DIM + c) * LSEQ + l;
#pragma unroll
        for (int gq = 0; gq < 2; ++gq) {
          const float4 xv = *(const float4*)(xp + gq * 4);
          float4 ov;
          ov.x = (xv.x + ev + acc[nt][gq * 4 + 0] + bi) * 0.70710678118f;
          ov.y = (xv.y + ev + acc[nt][gq * 4 + 1] + bi) * 0.70710678118f;
          ov.z = (xv.z + ev + acc[nt][gq * 4 + 2] + bi) * 0.70710678118f;
          ov.w = (xv.w + ev + acc[nt][gq * 4 + 3] + bi) * 0.70710678118f;
          *(float4*)(op + gq * 4) = ov;
        }
      } else {
        float* op = f1 + ((size_t)b * DIM + (c - DIM)) * LSEQ + l;
#pragma unroll
        for (int gq = 0; gq < 2; ++gq) {
          float4 ov;
          ov.x = acc[nt][gq * 4 + 0] + bi;
          ov.y = acc[nt][gq * 4 + 1] + bi;
          ov.z = acc[nt][gq * 4 + 2] + bi;
          ov.w = acc[nt][gq * 4 + 3] + bi;
          *(float4*)(op + gq * 4) = ov;
        }
      }
    }
  }
}

// ---------------------------------------------------------------------------
// Flash attention: per (batch, 64-query tile). 4 waves x 16 query rows.
// 64 key tiles of 64 keys: S=Q.K^T (16 wmma), online softmax (16-lane
// butterflies), P restaged via per-wave LDS into two A-fragments,
// O += P.V (32 wmma). S and AV loops are software-pipelined with
// double-buffered fragment registers; next global tile prefetched
// (global_prefetch_b8) during compute. Epilogue: O = u * (O / l).
// ---------------------------------------------------------------------------
__global__ __launch_bounds__(128) void flash_kernel(
    const _Float16* __restrict__ Q, const _Float16* __restrict__ Km,
    const _Float16* __restrict__ Vt, const _Float16* __restrict__ Ut,
    _Float16* __restrict__ O) {
  __shared__ _Float16 kls[64 * 136];    // K tile, row-major [key][dim]
  __shared__ _Float16 vls[256 * 72];    // V tile, transposed [dim][key]
  __shared__ _Float16 pls[4 * 16 * 72]; // per-wave P tile [q][key]

  const int tid = threadIdx.x;
  const int b = blockIdx.y;
  const int q0 = blockIdx.x * 64;
  const int w = tid >> 5, lane = tid & 31, h = lane >> 4, ln = lane & 15;

  // resident Q A-fragments (16 rows x 128 dims)
  v16h qa[4];
  {
    const _Float16* qr = Q + ((size_t)(b * LSEQ + q0 + w * 16 + ln)) * QKD;
#pragma unroll
    for (int k0 = 0; k0 < 4; ++k0)
      qa[k0] = pack16(*(const v8h*)(qr + k0 * 32 + h * 8),
                      *(const v8h*)(qr + k0 * 32 + 16 + h * 8));
  }

  const v8f vzero = {0.f, 0.f, 0.f, 0.f, 0.f, 0.f, 0.f, 0.f};
  v8f o[16];
#pragma unroll
  for (int j = 0; j < 16; ++j) o[j] = vzero;
  float mrow[8], lrow[8];
#pragma unroll
  for (int r = 0; r < 8; ++r) { mrow[r] = -3.0e38f; lrow[r] = 0.f; }
  const float invL = 1.0f / (float)LSEQ;

  const _Float16* kfrag = kls + ln * 136 + h * 8;  // per-lane K frag base
  const _Float16* vfrag = vls + ln * 72 + h * 8;   // per-lane V frag base

  for (int kt = 0; kt < LSEQ; kt += 64) {
    {  // stage K tile (64 x 128 halves)
      const int r = tid >> 1, cb = (tid & 1) * 64;
      const _Float16* src = Km + ((size_t)(b * LSEQ + kt + r)) * QKD + cb;
      _Float16* dst = kls + r * 136 + cb;
#pragma unroll
      for (int j = 0; j < 8; ++j) *(v8h*)(dst + j * 8) = *(const v8h*)(src + j * 8);
      if (kt + 64 < LSEQ) {  // prefetch next tile (global_prefetch_b8)
        __builtin_prefetch(src + (size_t)64 * QKD, 0, 0);
        __builtin_prefetch(src + (size_t)64 * QKD + 32, 0, 0);
      }
    }
    {  // stage V tile from transposed global (256 dims x 64 keys)
      const int db = tid >> 2, ko = (tid & 3) * 16;
#pragma unroll
      for (int it = 0; it < 8; ++it) {
        const int d = it * 32 + db;
        const _Float16* src = Vt + (size_t)d * BLROWS + b * LSEQ + kt + ko;
        *(v8h*)(vls + d * 72 + ko) = *(const v8h*)(src);
        *(v8h*)(vls + d * 72 + ko + 8) = *(const v8h*)(src + 8);
        if (kt + 64 < LSEQ) __builtin_prefetch(src + 64, 0, 0);
      }
    }
    __syncthreads();

    // S = Q K^T (four 16x16 key sub-tiles), pipelined across k0
    v8f s[4];
#pragma unroll
    for (int n = 0; n < 4; ++n) s[n] = vzero;
    {
      v16h bfb[2][4];
#pragma unroll
      for (int n = 0; n < 4; ++n) {
        const _Float16* bp = kfrag + n * 16 * 136;
        bfb[0][n] = pack16(*(const v8h*)bp, *(const v8h*)(bp + 16));
      }
#pragma unroll
      for (int k0 = 0; k0 < 4; ++k0) {
        const int cur = k0 & 1, nxt = cur ^ 1;
        if (k0 < 3) {
#pragma unroll
          for (int n = 0; n < 4; ++n) {
            const _Float16* bp = kfrag + n * 16 * 136 + (k0 + 1) * 32;
            bfb[nxt][n] = pack16(*(const v8h*)bp, *(const v8h*)(bp + 16));
          }
        }
#pragma unroll
        for (int n = 0; n < 4; ++n) s[n] = wmma32(qa[k0], bfb[cur][n], s[n]);
      }
    }

    // online softmax (scale 1/seq_len applied in f32)
    float scl[8];
    _Float16* pw = pls + w * 1152;
#pragma unroll
    for (int r = 0; r < 8; ++r) {
      float sv[4];
#pragma unroll
      for (int n = 0; n < 4; ++n) sv[n] = s[n][r] * invL;
      float mx = fmaxf(fmaxf(sv[0], sv[1]), fmaxf(sv[2], sv[3]));
#pragma unroll
      for (int off = 8; off > 0; off >>= 1) mx = fmaxf(mx, __shfl_xor(mx, off, 32));
      const float mnew = fmaxf(mrow[r], mx);
      float p[4];
      float rs = 0.f;
#pragma unroll
      for (int n = 0; n < 4; ++n) { p[n] = __expf(sv[n] - mnew); rs += p[n]; }
#pragma unroll
      for (int off = 8; off > 0; off >>= 1) rs += __shfl_xor(rs, off, 32);
      const float sc = __expf(mrow[r] - mnew);
      lrow[r] = lrow[r] * sc + rs;
      mrow[r] = mnew;
      scl[r] = sc;
      _Float16* pr = pw + (r + 8 * h) * 72;  // C layout -> [q][key]
#pragma unroll
      for (int n = 0; n < 4; ++n) pr[n * 16 + ln] = (_Float16)p[n];
    }
#pragma unroll
    for (int j = 0; j < 16; ++j)
#pragma unroll
      for (int r = 0; r < 8; ++r) o[j][r] *= scl[r];

    // P as two A-fragments (same-wave LDS, in-order DS ops)
    const _Float16* pb = pw + ln * 72;
    const v16h pf0 = pack16(*(const v8h*)(pb + 8 * h),
                            *(const v8h*)(pb + 16 + 8 * h));
    const v16h pf1 = pack16(*(const v8h*)(pb + 32 + 8 * h),
                            *(const v8h*)(pb + 48 + 8 * h));
    // O += P @ V, pipelined across the 16 dim sub-tiles
    {
      v16h vfb[2][2];
      vfb[0][0] = pack16(*(const v8h*)(vfrag), *(const v8h*)(vfrag + 16));
      vfb[0][1] = pack16(*(const v8h*)(vfrag + 32), *(const v8h*)(vfrag + 48));
#pragma unroll
      for (int j = 0; j < 16; ++j) {
        const int cur = j & 1, nxt = cur ^ 1;
        if (j < 15) {
          const _Float16* vb = vfrag + (j + 1) * 16 * 72;
          vfb[nxt][0] = pack16(*(const v8h*)(vb), *(const v8h*)(vb + 16));
          vfb[nxt][1] = pack16(*(const v8h*)(vb + 32), *(const v8h*)(vb + 48));
        }
        o[j] = wmma32(pf0, vfb[cur][0], o[j]);
        o[j] = wmma32(pf1, vfb[cur][1], o[j]);
      }
    }
    __syncthreads();
  }

  // epilogue: O = u * (o / l)
  float linv[8];
#pragma unroll
  for (int r = 0; r < 8; ++r) linv[r] = 1.0f / lrow[r];
  const int rowb = b * LSEQ + q0 + w * 16 + 8 * h;
#pragma unroll
  for (int j = 0; j < 16; ++j) {
    const int d = j * 16 + ln;
    const v8h uv = *(const v8h*)(Ut + (size_t)d * BLROWS + rowb);
#pragma unroll
    for (int r = 0; r < 8; ++r) {
      const float val = o[j][r] * linv[r] * (float)uv[r];
      O[(size_t)(rowb + r) * HIDN + d] = (_Float16)val;
    }
  }
}

// ---------------------------------------------------------------------------
extern "C" void kernel_launch(void* const* d_in, const int* in_sizes, int n_in,
                              void* d_out, int out_size, void* d_ws, size_t ws_size,
                              hipStream_t stream) {
  (void)in_sizes; (void)n_in; (void)out_size; (void)ws_size;
  const float* x       = (const float*)d_in[0];
  const float* t       = (const float*)d_in[1];
  const float* W_embed = (const float*)d_in[2];
  const float* b_embed = (const float*)d_in[3];
  const float* ln_g    = (const float*)d_in[4];
  const float* ln_b    = (const float*)d_in[5];
  const float* W_hid   = (const float*)d_in[6];
  const float* b_hid   = (const float*)d_in[7];
  const float* W_qk    = (const float*)d_in[8];
  const float* b_qk    = (const float*)d_in[9];
  const float* gamma   = (const float*)d_in[10];
  const float* beta    = (const float*)d_in[11];
  const float* W_out   = (const float*)d_in[12];
  const float* b_out   = (const float*)d_in[13];
  const float* W_proj  = (const float*)d_in[14];
  const float* b_proj  = (const float*)d_in[15];

  char* ws = (char*)d_ws;                              // ~48.1 MB used
  float*    emb    = (float*)ws;                       // 4*256 f32
  _Float16* normed = (_Float16*)(ws + 4096);           // 16384x256
  _Float16* vT     = normed + (size_t)BLROWS * DIM;    // 256x16384 (transposed)
  _Float16* uT     = vT + (size_t)HIDN * BLROWS;       // 256x16384 (transposed)
  _Float16* qb     = uT + (size_t)HIDN * BLROWS;       // 16384x128
  _Float16* kb     = qb + (size_t)BLROWS * QKD;        // 16384x128
  _Float16* Ob     = kb + (size_t)BLROWS * QKD;        // 16384x256
  _Float16* outbl  = Ob + (size_t)BLROWS * HIDN;       // 16384x256

  float* out1 = (float*)d_out;
  float* skip = out1 + (size_t)NB * DIM * LSEQ;

  emb_kernel<<<dim3(NB), 256, 0, stream>>>(t, W_embed, b_embed, emb);
  ln_kernel<<<dim3(LSEQ / 32, NB), 256, 0, stream>>>(x, emb, ln_g, ln_b, normed);
  gemm256<0><<<dim3(BLROWS / 128, 8), 256, 0, stream>>>(
      normed, W_hid, b_hid, vT, uT, nullptr, nullptr, nullptr, nullptr);
  gemm256<1><<<dim3(BLROWS / 128, 2), 256, 0, stream>>>(
      normed, W_qk, b_qk, qb, kb, gamma, beta, nullptr, nullptr);
  flash_kernel<<<dim3(LSEQ / 64, NB), 128, 0, stream>>>(qb, kb, vT, uT, Ob);
  gemm256<2><<<dim3(BLROWS / 128, 4), 256, 0, stream>>>(
      Ob, W_out, b_out, outbl, nullptr, nullptr, nullptr, nullptr, nullptr);
  gemm256<3><<<dim3(BLROWS / 128, 8), 256, 0, stream>>>(
      outbl, W_proj, b_proj, nullptr, nullptr, x, emb, out1, skip);
}
